// TLinear_19318762897738
// MI455X (gfx1250) — compile-verified, matrix operand order checked by reading, not run on hardware
//
#include <hip/hip_runtime.h>

#define D_INx   256
#define D_OUTx  256
#define NTOK    4096          // B*N = 4*1024
#define TG      64            // tokens per workgroup (2 pairs of 32)
#define NTG     (NTOK / TG)   // 64 token-groups
#define OS      64            // output rows (o) per workgroup
#define NOS     (D_OUTx / OS) // 4 o-slices
#define EPSc    1e-5f

typedef __attribute__((ext_vector_type(16))) __bf16 v16bf;
typedef __attribute__((ext_vector_type(8)))  __bf16 v8bf;
typedef __attribute__((ext_vector_type(4)))  __bf16 v4bf;
typedef __attribute__((ext_vector_type(8)))  float  v8f;
typedef __attribute__((ext_vector_type(4)))  float  v4f;

struct Halves { v8bf lo, hi; };
union ABFrag { v16bf v; Halves h; };

// ---------------- Wc fp32 -> bf16 (once; 33.5 MB result stays L2-resident) ------
__global__ void cvt_wc_bf16(const float* __restrict__ Wc, __bf16* __restrict__ Wb, int n4) {
    int i = blockIdx.x * blockDim.x + threadIdx.x;
    if (i >= n4) return;
    v4f f = ((const v4f*)Wc)[i];
    v4bf b = { (__bf16)f.x, (__bf16)f.y, (__bf16)f.z, (__bf16)f.w };
    ((v4bf*)Wb)[i] = b;
}

// Per-lane helper: build 8 loop-invariant B fragments (one 16-token tile)
__device__ __forceinline__ void build_B(const float* __restrict__ feats,
                                        int t, int half, ABFrag Bf[8]) {
#pragma unroll
    for (int k = 0; k < 8; ++k) {
        const float* fp = feats + (size_t)t * D_INx + k * 32 + half * 8;
        v8bf lo, hi;
#pragma unroll
        for (int e = 0; e < 8; ++e) {
            lo[e] = (__bf16)fp[e];
            hi[e] = (__bf16)fp[16 + e];
        }
        Bf[k].h.lo = lo; Bf[k].h.hi = hi;
    }
}

// ---------------- Kernel A: dsum[t][i] += sum over this WG's o-slice of |wc| ----
__launch_bounds__(256)
__global__ void tlinear_dsum(const float* __restrict__ feats,
                             const __bf16* __restrict__ Wb,
                             const float* __restrict__ bc,
                             float* __restrict__ dsum) {
    const int lane  = threadIdx.x & 31;
    const int wid   = threadIdx.x >> 5;
    const int half  = lane >> 4;
    const int tl    = lane & 15;
    const int pair  = wid & 1;                    // which 32-token pair
    const int osub  = wid >> 1;                   // which 16-o subrange
    const int tA    = blockIdx.x * TG + pair * 32 + tl;
    const int tB    = tA + 16;
    const int obase = blockIdx.y * OS + osub * 16;
    const int r     = half * 8;

    ABFrag Bf0[8], Bf1[8];                        // two B-tile sets: 128 VGPRs
    build_B(feats, tA, half, Bf0);
    build_B(feats, tB, half, Bf1);

    for (int it = 0; it < 16; ++it) {
        float acc0[8] = {0.f,0.f,0.f,0.f,0.f,0.f,0.f,0.f};
        float acc1[8] = {0.f,0.f,0.f,0.f,0.f,0.f,0.f,0.f};
        for (int oo = 0; oo < 16; ++oo) {
            const int rowBase = (obase + oo) * 256 + it * 16;
            v4f b0 = *(const v4f*)(bc + rowBase + r);
            v4f b1 = *(const v4f*)(bc + rowBase + r + 4);
            v8f D0 = { b0.x, b0.y, b0.z, b0.w, b1.x, b1.y, b1.z, b1.w };
            v8f D1 = D0;                          // same bias rows for both tiles
            const __bf16* ap = Wb + (size_t)(rowBase + tl) * D_INx;
#pragma unroll
            for (int k = 0; k < 8; ++k) {
                ABFrag A;                         // each A feeds TWO wmmas
                A.h.lo = *(const v8bf*)(ap + k * 32 + half * 8);
                A.h.hi = *(const v8bf*)(ap + k * 32 + 16 + half * 8);
                D0 = __builtin_amdgcn_wmma_f32_16x16x32_bf16(
                         false, A.v, false, Bf0[k].v, (short)0, D0, false, false);
                D1 = __builtin_amdgcn_wmma_f32_16x16x32_bf16(
                         false, A.v, false, Bf1[k].v, (short)0, D1, false, false);
            }
#pragma unroll
            for (int j = 0; j < 8; ++j) {
                acc0[j] += __builtin_fabsf(D0[j]);
                acc1[j] += __builtin_fabsf(D1[j]);
            }
        }
#pragma unroll
        for (int j = 0; j < 8; ++j) {
            unsafeAtomicAdd(&dsum[(size_t)tA * D_INx + it * 16 + r + j], acc0[j]);
            unsafeAtomicAdd(&dsum[(size_t)tB * D_INx + it * 16 + r + j], acc1[j]);
        }
    }
}

// ---------------- Kernel B: g[t][i] = f[t][i] / (dsum[t][i] + 256*eps) ----------
__global__ void tlinear_g(const float* __restrict__ feats,
                          const float* __restrict__ dsum,
                          float* __restrict__ g, int n4) {
    int i = blockIdx.x * blockDim.x + threadIdx.x;
    if (i >= n4) return;
    v4f f = ((const v4f*)feats)[i];
    v4f d = ((const v4f*)dsum)[i];
    const float e = (float)D_OUTx * EPSc;
    v4f o = { f.x / (d.x + e), f.y / (d.y + e), f.z / (d.z + e), f.w / (d.w + e) };
    ((v4f*)g)[i] = o;
}

// ---------------- Kernel C: out[t][o] = sum_i wc[t,o,i] * g[t][i] ---------------
__launch_bounds__(256)
__global__ void tlinear_out(const float* __restrict__ feats,
                            const __bf16* __restrict__ Wb,
                            const float* __restrict__ bc,
                            const float* __restrict__ g,
                            float* __restrict__ out) {
    const int lane  = threadIdx.x & 31;
    const int wid   = threadIdx.x >> 5;
    const int half  = lane >> 4;
    const int tl    = lane & 15;
    const int pair  = wid & 1;
    const int osub  = wid >> 1;
    const int tA    = blockIdx.x * TG + pair * 32 + tl;
    const int tB    = tA + 16;
    const int obase = blockIdx.y * OS + osub * 16;
    const int r     = half * 8;

    ABFrag Bf0[8], Bf1[8];
    build_B(feats, tA, half, Bf0);
    build_B(feats, tB, half, Bf1);

    for (int oo = 0; oo < 16; ++oo) {
        const int o = obase + oo;
        float sum0 = 0.f, sum1 = 0.f;
        for (int it = 0; it < 16; ++it) {
            const int rowBase = o * 256 + it * 16;
            v4f b0 = *(const v4f*)(bc + rowBase + r);
            v4f b1 = *(const v4f*)(bc + rowBase + r + 4);
            v8f D0 = { b0.x, b0.y, b0.z, b0.w, b1.x, b1.y, b1.z, b1.w };
            v8f D1 = D0;
            const __bf16* ap = Wb + (size_t)(rowBase + tl) * D_INx;
#pragma unroll
            for (int k = 0; k < 8; ++k) {
                ABFrag A;
                A.h.lo = *(const v8bf*)(ap + k * 32 + half * 8);
                A.h.hi = *(const v8bf*)(ap + k * 32 + 16 + half * 8);
                D0 = __builtin_amdgcn_wmma_f32_16x16x32_bf16(
                         false, A.v, false, Bf0[k].v, (short)0, D0, false, false);
                D1 = __builtin_amdgcn_wmma_f32_16x16x32_bf16(
                         false, A.v, false, Bf1[k].v, (short)0, D1, false, false);
            }
            const float* gpA = g + (size_t)tA * D_INx + it * 16 + r;  // L0-hot
            const float* gpB = g + (size_t)tB * D_INx + it * 16 + r;
            v4f a0 = *(const v4f*)gpA;
            v4f a1 = *(const v4f*)(gpA + 4);
            v4f c0 = *(const v4f*)gpB;
            v4f c1 = *(const v4f*)(gpB + 4);
            sum0 += D0[0]*a0.x + D0[1]*a0.y + D0[2]*a0.z + D0[3]*a0.w
                  + D0[4]*a1.x + D0[5]*a1.y + D0[6]*a1.z + D0[7]*a1.w;
            sum1 += D1[0]*c0.x + D1[1]*c0.y + D1[2]*c0.z + D1[3]*c0.w
                  + D1[4]*c1.x + D1[5]*c1.y + D1[6]*c1.z + D1[7]*c1.w;
        }
        sum0 += __shfl_xor(sum0, 16, 32);   // fold the two i%16 halves
        sum1 += __shfl_xor(sum1, 16, 32);
        if (half == 0) {
            out[(size_t)tA * D_OUTx + o] = sum0;
            out[(size_t)tB * D_OUTx + o] = sum1;
        }
    }
}

extern "C" void kernel_launch(void* const* d_in, const int* in_sizes, int n_in,
                              void* d_out, int out_size, void* d_ws, size_t ws_size,
                              hipStream_t stream) {
    const float* feats = (const float*)d_in[0];   // (4,1024,256) fp32
    const float* Wc    = (const float*)d_in[1];   // (65536,256)  fp32
    const float* bc    = (const float*)d_in[2];   // (65536,)     fp32
    float* out = (float*)d_out;                   // (4,1024,256) fp32

    // workspace layout: [ Wc-bf16 : 33.5MB ][ dsum : 4MB ][ g : 4MB ]
    __bf16* Wb  = (__bf16*)d_ws;
    float* dsum = (float*)((char*)d_ws + (size_t)D_OUTx * D_INx * D_INx * 2);
    float* gbuf = dsum + (size_t)NTOK * D_INx;

    const int nWc4 = (D_OUTx * D_INx * D_INx) / 4;
    cvt_wc_bf16<<<(nWc4 + 255) / 256, 256, 0, stream>>>(Wc, Wb, nWc4);
    hipMemsetAsync(dsum, 0, (size_t)NTOK * D_INx * sizeof(float), stream);

    dim3 grid(NTG, NOS);  // 64 x 4 = 256 workgroups
    tlinear_dsum<<<grid, 256, 0, stream>>>(feats, Wb, bc, dsum);

    const int nG4 = (NTOK * D_INx) / 4;
    tlinear_g<<<(nG4 + 255) / 256, 256, 0, stream>>>(feats, dsum, gbuf, nG4);

    tlinear_out<<<grid, 256, 0, stream>>>(feats, Wb, bc, gbuf, out);
}
